// MultiHeadAttention2_72885595013339
// MI455X (gfx1250) — matrix operand order repaired and checked by reading, not compiled, not run on hardware
//
#include <hip/hip_runtime.h>
#include <hip/hip_bf16.h>

typedef __attribute__((ext_vector_type(16))) __bf16 v16bf;
typedef __attribute__((ext_vector_type(8)))  __bf16 v8bf;
typedef __attribute__((ext_vector_type(4)))  __bf16 v4bf;
typedef __attribute__((ext_vector_type(8)))  float  v8f;

#define S_LEN 2048
#define D_DIM 1024
#define NH    16
#define HDIM  64

static __device__ __forceinline__ v8f wmma_bf16(v16bf a, v16bf b, v8f c) {
  return __builtin_amdgcn_wmma_f32_16x16x32_bf16(false, a, false, b, (short)0, c,
                                                 false, false);
}
static __device__ __forceinline__ v16bf join8(v8bf lo, v8bf hi) {
  return __builtin_shufflevector(lo, hi, 0, 1, 2, 3, 4, 5, 6, 7,
                                         8, 9, 10, 11, 12, 13, 14, 15);
}
// gfx1250 async global->LDS copy (16B per lane, ASYNCcnt-tracked)
static __device__ __forceinline__ void async_copy16(unsigned lds_off, const void* g) {
  asm volatile("global_load_async_to_lds_b128 %0, %1, off"
               :: "v"(lds_off), "v"((unsigned long long)(size_t)g)
               : "memory");
}
static __device__ __forceinline__ void wait_async0() {
  asm volatile("s_wait_asynccnt 0" ::: "memory");
}

// ------------------------------------- fp32 -> bf16, vectorized (n4 = n/4 chunks)
__global__ __launch_bounds__(256) void cvt_bf16v_kernel(const float* __restrict__ src,
                                                        __bf16* __restrict__ dst, int n4) {
  int i = blockIdx.x * 256 + threadIdx.x;
  if (i < n4) {
    float4 f = ((const float4*)src)[i];
    v4bf p;
    p[0] = (__bf16)f.x; p[1] = (__bf16)f.y; p[2] = (__bf16)f.z; p[3] = (__bf16)f.w;
    ((v4bf*)dst)[i] = p;
  }
}

// ----------------------------------------- fp32 [K,N] -> bf16 transposed [N,K]
__global__ __launch_bounds__(256) void cvt_t_kernel(const float* __restrict__ src,
                                                    __bf16* __restrict__ dst) {
  __shared__ __bf16 t[32][33];
  const int tx = threadIdx.x & 31, ty = threadIdx.x >> 5;
  const int k0 = blockIdx.y * 32, n0 = blockIdx.x * 32;
#pragma unroll
  for (int i = 0; i < 4; ++i) {
    int y = ty + i * 8;
    t[y][tx] = (__bf16)src[(size_t)(k0 + y) * 1024 + n0 + tx];
  }
  __syncthreads();
#pragma unroll
  for (int i = 0; i < 4; ++i) {
    int y = ty + i * 8;
    dst[(size_t)(n0 + y) * 1024 + k0 + tx] = t[tx][y];
  }
}

// ---------------- QKV projection GEMM: bf16 WMMA, double-buffered async DMA staging
// Xb: [4096,1024] bf16 ; Wt: [N][K] bf16 (pre-transposed) ; bias fp32
// vtrans==0: Out[b,h,s,hd] ; vtrans==1: Out[b,h,hd,s]
__global__ __launch_bounds__(256) void qkv_proj_kernel(const __bf16* __restrict__ Xb,
                                                       const __bf16* __restrict__ Wt,
                                                       const float* __restrict__ bias,
                                                       __bf16* __restrict__ Out,
                                                       int vtrans) {
  __shared__ __bf16 sA[2][128][40];    // M x K(32), 80B row stride, double-buffered
  __shared__ __bf16 sBt[2][128][40];   // N x K(32)

  const int tid  = threadIdx.x;
  const int wave = tid >> 5;
  const int lane = tid & 31;
  const int lr   = lane & 15;
  const int hi   = lane >> 4;
  const int bm   = blockIdx.y * 128;
  const int bn   = blockIdx.x * 128;
  const int wm   = (wave & 1) * 64;
  const int wn   = (wave >> 1) * 32;

  auto stage = [&](int kc, int bufi) {
    const int k0 = kc * 32;
#pragma unroll
    for (int i = 0; i < 2; ++i) {
      int idx = tid + i * 256;
      int r = idx >> 2, c8 = (idx & 3) * 8;
      async_copy16((unsigned)(size_t)&sA[bufi][r][c8],
                   Xb + (size_t)(bm + r) * 1024 + k0 + c8);
      async_copy16((unsigned)(size_t)&sBt[bufi][r][c8],
                   Wt + (size_t)(bn + r) * 1024 + k0 + c8);
    }
  };

  v8f acc[4][2];
#pragma unroll
  for (int i = 0; i < 4; ++i)
#pragma unroll
    for (int j = 0; j < 2; ++j) acc[i][j] = (v8f){0.f,0.f,0.f,0.f,0.f,0.f,0.f,0.f};

  stage(0, 0);
  for (int kc = 0; kc < 32; ++kc) {
    const int bufi = kc & 1;
    wait_async0();
    __syncthreads();                 // compiler adds s_wait_dscnt: prior reads done
    if (kc + 1 < 32) stage(kc + 1, bufi ^ 1);   // DMA next slab during compute

    v16bf afrag[4];
#pragma unroll
    for (int i = 0; i < 4; ++i) {
      int row = wm + 16 * i + lr;
      v8bf lo = *(const v8bf*)&sA[bufi][row][hi * 16];
      v8bf hi8 = *(const v8bf*)&sA[bufi][row][hi * 16 + 8];
      afrag[i] = join8(lo, hi8);
    }
    v16bf bfrag[2];
#pragma unroll
    for (int j = 0; j < 2; ++j) {
      int col = wn + 16 * j + lr;
      v8bf lo = *(const v8bf*)&sBt[bufi][col][hi * 16];
      v8bf hi8 = *(const v8bf*)&sBt[bufi][col][hi * 16 + 8];
      bfrag[j] = join8(lo, hi8);
    }
#pragma unroll
    for (int i = 0; i < 4; ++i)
#pragma unroll
      for (int j = 0; j < 2; ++j)
        acc[i][j] = wmma_bf16(afrag[i], bfrag[j], acc[i][j]);
  }

#pragma unroll
  for (int i = 0; i < 4; ++i)
#pragma unroll
    for (int j = 0; j < 2; ++j)
#pragma unroll
      for (int p = 0; p < 8; ++p) {
        int r = bm + wm + 16 * i + hi * 8 + p;     // b*S + s
        int c = bn + wn + 16 * j + lr;             // h*64 + hd
        float v = acc[i][j][p] + bias[c];
        int b = r >> 11, s = r & 2047, h = c >> 6, hd = c & 63;
        size_t idx = vtrans
            ? ((((size_t)b * NH + h) * HDIM + hd) * S_LEN + s)   // [B,H,hd,S]
            : ((((size_t)b * NH + h) * S_LEN + s) * HDIM + hd);  // [B,H,S,hd]
        Out[idx] = (__bf16)v;
      }
}

// ------------------------------------------- flash-style masked attention (WMMA)
// qb/kb: [B,H,S,64] bf16 ; vt: [B,H,64,S] bf16 ; mask: [B,S,S] u8 (nonzero=masked)
__global__ __launch_bounds__(256) void attn_kernel(const __bf16* __restrict__ qb,
                                                   const __bf16* __restrict__ kb,
                                                   const __bf16* __restrict__ vt,
                                                   const unsigned char* __restrict__ mask,
                                                   __bf16* __restrict__ ctx) {
  __shared__ __bf16 sP[8][16][40];   // per-wave 16x32 P tile, 80B row stride

  const int tid  = threadIdx.x;
  const int wave = tid >> 5;
  const int lane = tid & 31;
  const int lr   = lane & 15;
  const int hi   = lane >> 4;
  const int h    = blockIdx.y;
  const int b    = blockIdx.z;
  const int bh   = b * NH + h;
  const int q0   = blockIdx.x * 128 + wave * 16;

  const __bf16* qh = qb + (size_t)bh * S_LEN * HDIM;
  const __bf16* kh = kb + (size_t)bh * S_LEN * HDIM;
  const __bf16* vh = vt + (size_t)bh * HDIM * S_LEN;   // [64][2048]
  const unsigned char* mb = mask + (size_t)b * S_LEN * S_LEN;

  v16bf aq[2];
#pragma unroll
  for (int ks = 0; ks < 2; ++ks) {
    int base = ks * 32 + hi * 16;
    int row = q0 + lr;
    v8bf lo = *(const v8bf*)(qh + (size_t)row * HDIM + base);
    v8bf hi8 = *(const v8bf*)(qh + (size_t)row * HDIM + base + 8);
    aq[ks] = join8(lo, hi8);
  }

  v8f o[4];
#pragma unroll
  for (int t = 0; t < 4; ++t) o[t] = (v8f){0.f,0.f,0.f,0.f,0.f,0.f,0.f,0.f};
  float m[8], l[8];
#pragma unroll
  for (int p = 0; p < 8; ++p) { m[p] = -3.0e38f; l[p] = 0.f; }

  for (int key0 = 0; key0 < S_LEN; key0 += 32) {
    v8f s[2];
#pragma unroll
    for (int j = 0; j < 2; ++j) {
      s[j] = (v8f){0.f,0.f,0.f,0.f,0.f,0.f,0.f,0.f};
      int col = key0 + j * 16 + lr;
#pragma unroll
      for (int ks = 0; ks < 2; ++ks) {
        v16bf bk = *(const v16bf*)(kh + (size_t)col * HDIM + ks * 32 + hi * 16);
        s[j] = wmma_bf16(aq[ks], bk, s[j]);
      }
    }
    float ps[2][8], tmax[8];
#pragma unroll
    for (int p = 0; p < 8; ++p) tmax[p] = -3.0e38f;
#pragma unroll
    for (int j = 0; j < 2; ++j)
#pragma unroll
      for (int p = 0; p < 8; ++p) {
        int row = q0 + hi * 8 + p;
        int key = key0 + j * 16 + lr;
        float sv = s[j][p] * 0.125f;
        if (mb[(size_t)row * S_LEN + key]) sv = -1.0e9f;
        ps[j][p] = sv;
        tmax[p] = fmaxf(tmax[p], sv);
      }
#pragma unroll
    for (int xm = 8; xm >= 1; xm >>= 1)
#pragma unroll
      for (int p = 0; p < 8; ++p)
        tmax[p] = fmaxf(tmax[p], __shfl_xor(tmax[p], xm, 32));
    float corr[8], rowsum[8];
#pragma unroll
    for (int p = 0; p < 8; ++p) {
      float nm = fmaxf(m[p], tmax[p]);
      corr[p] = __expf(m[p] - nm);
      m[p] = nm;
      rowsum[p] = 0.f;
    }
#pragma unroll
    for (int j = 0; j < 2; ++j)
#pragma unroll
      for (int p = 0; p < 8; ++p) {
        float e = __expf(ps[j][p] - m[p]);
        ps[j][p] = e;
        rowsum[p] += e;
      }
#pragma unroll
    for (int xm = 8; xm >= 1; xm >>= 1)
#pragma unroll
      for (int p = 0; p < 8; ++p)
        rowsum[p] += __shfl_xor(rowsum[p], xm, 32);
#pragma unroll
    for (int p = 0; p < 8; ++p) l[p] = l[p] * corr[p] + rowsum[p];
#pragma unroll
    for (int t = 0; t < 4; ++t)
#pragma unroll
      for (int p = 0; p < 8; ++p) o[t][p] *= corr[p];

    // P: C-layout -> A-layout via wave-private LDS (same-wave DS in-order)
#pragma unroll
    for (int j = 0; j < 2; ++j)
#pragma unroll
      for (int p = 0; p < 8; ++p)
        sP[wave][hi * 8 + p][j * 16 + lr] = (__bf16)ps[j][p];
    v16bf ap;
    {
      v8bf lo = *(const v8bf*)&sP[wave][lr][hi * 16];
      v8bf hi8 = *(const v8bf*)&sP[wave][lr][hi * 16 + 8];
      ap = join8(lo, hi8);
    }
#pragma unroll
    for (int t = 0; t < 4; ++t) {
      int hd = t * 16 + lr;
      v16bf bv = *(const v16bf*)(vh + (size_t)hd * S_LEN + key0 + hi * 16);
      o[t] = wmma_bf16(ap, bv, o[t]);
    }
  }

  float inv[8];
#pragma unroll
  for (int p = 0; p < 8; ++p) inv[p] = 1.0f / l[p];
#pragma unroll
  for (int t = 0; t < 4; ++t)
#pragma unroll
    for (int p = 0; p < 8; ++p) {
      int row = q0 + hi * 8 + p;
      int hd  = t * 16 + lr;
      ctx[((size_t)b * S_LEN + row) * D_DIM + h * HDIM + hd] = (__bf16)(o[t][p] * inv[p]);
    }
}

// ------------ output projection + bias + residual (double-buffered async staging)
__global__ __launch_bounds__(256) void out_proj_kernel(const __bf16* __restrict__ Xc,
                                                       const __bf16* __restrict__ Wt,
                                                       const float* __restrict__ bias,
                                                       const float* __restrict__ residual,
                                                       float* __restrict__ Out) {
  __shared__ __bf16 sA[2][128][40];
  __shared__ __bf16 sBt[2][128][40];

  const int tid  = threadIdx.x;
  const int wave = tid >> 5;
  const int lane = tid & 31;
  const int lr   = lane & 15;
  const int hi   = lane >> 4;
  const int bm   = blockIdx.y * 128;
  const int bn   = blockIdx.x * 128;
  const int wm   = (wave & 1) * 64;
  const int wn   = (wave >> 1) * 32;

  auto stage = [&](int kc, int bufi) {
    const int k0 = kc * 32;
#pragma unroll
    for (int i = 0; i < 2; ++i) {
      int idx = tid + i * 256;
      int r = idx >> 2, c8 = (idx & 3) * 8;
      async_copy16((unsigned)(size_t)&sA[bufi][r][c8],
                   Xc + (size_t)(bm + r) * 1024 + k0 + c8);
      async_copy16((unsigned)(size_t)&sBt[bufi][r][c8],
                   Wt + (size_t)(bn + r) * 1024 + k0 + c8);
    }
  };

  v8f acc[4][2];
#pragma unroll
  for (int i = 0; i < 4; ++i)
#pragma unroll
    for (int j = 0; j < 2; ++j) acc[i][j] = (v8f){0.f,0.f,0.f,0.f,0.f,0.f,0.f,0.f};

  stage(0, 0);
  for (int kc = 0; kc < 32; ++kc) {
    const int bufi = kc & 1;
    wait_async0();
    __syncthreads();
    if (kc + 1 < 32) stage(kc + 1, bufi ^ 1);

    v16bf afrag[4];
#pragma unroll
    for (int i = 0; i < 4; ++i) {
      int row = wm + 16 * i + lr;
      v8bf lo = *(const v8bf*)&sA[bufi][row][hi * 16];
      v8bf hi8 = *(const v8bf*)&sA[bufi][row][hi * 16 + 8];
      afrag[i] = join8(lo, hi8);
    }
    v16bf bfrag[2];
#pragma unroll
    for (int j = 0; j < 2; ++j) {
      int col = wn + 16 * j + lr;
      v8bf lo = *(const v8bf*)&sBt[bufi][col][hi * 16];
      v8bf hi8 = *(const v8bf*)&sBt[bufi][col][hi * 16 + 8];
      bfrag[j] = join8(lo, hi8);
    }
#pragma unroll
    for (int i = 0; i < 4; ++i)
#pragma unroll
      for (int j = 0; j < 2; ++j)
        acc[i][j] = wmma_bf16(afrag[i], bfrag[j], acc[i][j]);
  }

#pragma unroll
  for (int i = 0; i < 4; ++i)
#pragma unroll
    for (int j = 0; j < 2; ++j)
#pragma unroll
      for (int p = 0; p < 8; ++p) {
        int r = bm + wm + 16 * i + hi * 8 + p;
        int c = bn + wn + 16 * j + lr;
        size_t idx = (size_t)r * 1024 + c;
        Out[idx] = acc[i][j][p] + bias[c] + residual[idx];
      }
}

// ------------------------------------------------------------------- LayerNorm
__global__ __launch_bounds__(256) void ln_kernel(const float* __restrict__ X,
                                                 const float* __restrict__ gamma,
                                                 const float* __restrict__ beta,
                                                 float* __restrict__ out) {
  __shared__ float ssum[8], ssq[8];
  const int tid = threadIdx.x;
  const size_t base = (size_t)blockIdx.x * D_DIM;
  float vals[4];
  float sum = 0.f, sq = 0.f;
#pragma unroll
  for (int i = 0; i < 4; ++i) {
    float v = X[base + tid + i * 256];
    vals[i] = v; sum += v; sq += v * v;
  }
#pragma unroll
  for (int xm = 16; xm >= 1; xm >>= 1) {
    sum += __shfl_xor(sum, xm, 32);
    sq  += __shfl_xor(sq,  xm, 32);
  }
  if ((tid & 31) == 0) { ssum[tid >> 5] = sum; ssq[tid >> 5] = sq; }
  __syncthreads();
  float ts = 0.f, tq = 0.f;
#pragma unroll
  for (int wv = 0; wv < 8; ++wv) { ts += ssum[wv]; tq += ssq[wv]; }
  const float mu = ts * (1.0f / D_DIM);
  const float var = tq * (1.0f / D_DIM) - mu * mu;
  const float rstd = rsqrtf(var + 1e-5f);
#pragma unroll
  for (int i = 0; i < 4; ++i) {
    int c = tid + i * 256;
    out[base + c] = (vals[i] - mu) * rstd * gamma[c] + beta[c];
  }
}

// ------------------------------------------------------------------- launcher
extern "C" void kernel_launch(void* const* d_in, const int* in_sizes, int n_in,
                              void* d_out, int out_size, void* d_ws, size_t ws_size,
                              hipStream_t stream) {
  (void)in_sizes; (void)n_in; (void)out_size; (void)ws_size;
  const float* Q  = (const float*)d_in[0];
  const float* K  = (const float*)d_in[1];
  const float* V  = (const float*)d_in[2];
  const unsigned char* mask = (const unsigned char*)d_in[3];
  const float* Wq = (const float*)d_in[4];
  const float* bq = (const float*)d_in[5];
  const float* Wk = (const float*)d_in[6];
  const float* bk = (const float*)d_in[7];
  const float* Wv = (const float*)d_in[8];
  const float* bv = (const float*)d_in[9];
  const float* Wo = (const float*)d_in[10];
  const float* bo = (const float*)d_in[11];
  const float* gamma = (const float*)d_in[12];
  const float* beta  = (const float*)d_in[13];

  char* w = (char*)d_ws;
  __bf16* Wqt = (__bf16*)(w + (size_t) 0 * (1u << 20));  // 2MB each, [N][K] bf16
  __bf16* Wkt = (__bf16*)(w + (size_t) 2 * (1u << 20));
  __bf16* Wvt = (__bf16*)(w + (size_t) 4 * (1u << 20));
  __bf16* Wot = (__bf16*)(w + (size_t) 6 * (1u << 20));
  __bf16* qb  = (__bf16*)(w + (size_t) 8 * (1u << 20));  // 8MB [B,H,S,64]
  __bf16* kb  = (__bf16*)(w + (size_t)16 * (1u << 20));  // 8MB [B,H,S,64]
  __bf16* vt  = (__bf16*)(w + (size_t)24 * (1u << 20));  // 8MB [B,H,64,S]
  __bf16* ctx = (__bf16*)(w + (size_t)32 * (1u << 20));  // 8MB [B,S,1024]
  float*  xbuf= (float*) (w + (size_t)40 * (1u << 20));  // 16MB
  __bf16* qxb = (__bf16*)(w + (size_t)56 * (1u << 20));  // 8MB [4096,1024] bf16
  __bf16* kxb = (__bf16*)(w + (size_t)64 * (1u << 20));
  __bf16* vxb = (__bf16*)(w + (size_t)72 * (1u << 20));

  dim3 tgrid(32, 32);
  cvt_t_kernel<<<tgrid, 256, 0, stream>>>(Wq, Wqt);
  cvt_t_kernel<<<tgrid, 256, 0, stream>>>(Wk, Wkt);
  cvt_t_kernel<<<tgrid, 256, 0, stream>>>(Wv, Wvt);
  cvt_t_kernel<<<tgrid, 256, 0, stream>>>(Wo, Wot);

  const int N4 = 4096 * 1024 / 4;
  cvt_bf16v_kernel<<<N4 / 256, 256, 0, stream>>>(Q, qxb, N4);
  cvt_bf16v_kernel<<<N4 / 256, 256, 0, stream>>>(K, kxb, N4);
  cvt_bf16v_kernel<<<N4 / 256, 256, 0, stream>>>(V, vxb, N4);

  dim3 gemm_grid(1024 / 128, 4096 / 128);
  qkv_proj_kernel<<<gemm_grid, 256, 0, stream>>>(qxb, Wqt, bq, qb, 0);
  qkv_proj_kernel<<<gemm_grid, 256, 0, stream>>>(kxb, Wkt, bk, kb, 0);
  qkv_proj_kernel<<<gemm_grid, 256, 0, stream>>>(vxb, Wvt, bv, vt, 1);

  attn_kernel<<<dim3(S_LEN / 128, NH, 2), 256, 0, stream>>>(qb, kb, vt, mask, ctx);

  out_proj_kernel<<<gemm_grid, 256, 0, stream>>>(ctx, Wot, bo, Q, xbuf);

  ln_kernel<<<4096, 256, 0, stream>>>(xbuf, gamma, beta, (float*)d_out);
}
